// LionCrossAttentionDimH_85341000171879
// MI455X (gfx1250) — matrix operand order repaired and checked by reading, not compiled, not run on hardware
//
#include <hip/hip_runtime.h>
#include <hip/hip_bf16.h>
#include <math.h>

// ---------------------------------------------------------------------------
// LionCrossAttentionDimH for MI455X (gfx1250, wave32, WMMA).
// All GEMMs use v_wmma_f32_16x16x32_f16 with f32 accumulation.
// Fragments are K-contiguous per lane -> two ds_load_b128 each.
// Staging is straight-line: batched global_load_b128 before ds_store_b128.
// GEMM epilogues are template-specialized (no per-element branching).
// ---------------------------------------------------------------------------

typedef __attribute__((ext_vector_type(16))) _Float16 v16h;
typedef __attribute__((ext_vector_type(8)))  _Float16 v8h;
typedef __attribute__((ext_vector_type(8)))  float    v8f;
typedef __attribute__((ext_vector_type(4)))  float    v4f;

// D = A(16x32 f16) * B(32x16 f16) + C(16x16 f32)
__device__ __forceinline__ v8f wmma_16x16x32(v16h a, v16h b, v8f c) {
  return __builtin_amdgcn_wmma_f32_16x16x32_f16(
      /*neg_a=*/false, a, /*neg_b=*/false, b,
      /*c_mod=*/(short)0, c, /*reuse_a=*/false, /*reuse_b=*/false);
}

// A fragment (16x32 f16): LDS tile row-major [M][K], ld halves/row (16B-aligned
// rows).  lane l: M=l&15, K=(l>>4)*16 + 0..15 -> contiguous.
__device__ __forceinline__ v16h frag_a(const _Float16* tile, int ld, int lane) {
  const _Float16* p = tile + (lane & 15) * ld + ((lane >> 4) << 4);
  v8h lo = *(const v8h*)p;
  v8h hi = *(const v8h*)(p + 8);
  return __builtin_shufflevector(lo, hi, 0, 1, 2, 3, 4, 5, 6, 7,
                                 8, 9, 10, 11, 12, 13, 14, 15);
}

// B fragment (32x16 f16) from transposed tile B^T [N][K] in LDS.
// lane l: N=l&15, K=(l>>4)*16 + 0..15 -> contiguous.
__device__ __forceinline__ v16h frag_bt(const _Float16* tile, int ld, int lane) {
  const _Float16* p = tile + (lane & 15) * ld + ((lane >> 4) << 4);
  v8h lo = *(const v8h*)p;
  v8h hi = *(const v8h*)(p + 8);
  return __builtin_shufflevector(lo, hi, 0, 1, 2, 3, 4, 5, 6, 7,
                                 8, 9, 10, 11, 12, 13, 14, 15);
}

// ---------------------------------------------------------------------------
// LayerNorm over last dim D (compile-time), output f16.  One block per row.
// ---------------------------------------------------------------------------
template <int D>
__global__ void __launch_bounds__(256)
ln_cast_kernel(const float* __restrict__ x, const float* __restrict__ g,
               const float* __restrict__ b, _Float16* __restrict__ out) {
  const int row = blockIdx.x;
  const float* xr = x + (size_t)row * D;
  float s = 0.f, ss = 0.f;
#pragma unroll
  for (int i = 0; i < D / 256; ++i) {
    float v = xr[threadIdx.x + i * 256];
    s += v; ss += v * v;
  }
  __shared__ float r1[256], r2[256];
  r1[threadIdx.x] = s; r2[threadIdx.x] = ss;
  __syncthreads();
  for (int off = 128; off > 0; off >>= 1) {
    if (threadIdx.x < off) { r1[threadIdx.x] += r1[threadIdx.x + off];
                             r2[threadIdx.x] += r2[threadIdx.x + off]; }
    __syncthreads();
  }
  const float mean = r1[0] * (1.0f / D);
  const float var  = r2[0] * (1.0f / D) - mean * mean;
  const float inv  = rsqrtf(var + 1e-5f);
#pragma unroll
  for (int i = 0; i < D / 256; ++i) {
    const int idx = threadIdx.x + i * 256;
    float v = (xr[idx] - mean) * inv * g[idx] + b[idx];
    out[(size_t)row * D + idx] = (_Float16)v;
  }
}

// Convert W[K,N] f32 -> W^T[N,K] f16.  grid = (N/256, K).
__global__ void __launch_bounds__(256)
cvt_transpose_kernel(const float* __restrict__ in, _Float16* __restrict__ out,
                     int K, int N) {
  const int n = blockIdx.x * 256 + threadIdx.x;
  const int k = blockIdx.y;
  if (n < N) out[(size_t)n * K + k] = (_Float16)in[(size_t)k * N + n];
}

// ---------------------------------------------------------------------------
// C[M,N] = A[M,K](f16) * W(f16 given as W^T[N,K]) + bias.
// 256 thr = 8 waves, tile 128x128, K-step 32, wave tile 32x64.
// OMODE 0: f16 out, row-major [M,N]          (dst16)
// OMODE 1: f16 out, [bh][col][48] batch-transposed (bh=row/48, w=row%48)
// OMODE 2: f32 out, row-major, + residual    (dst32, residual)
// ---------------------------------------------------------------------------
template <int OMODE>
__global__ void __launch_bounds__(256)
gemm_f16_kernel(const _Float16* __restrict__ A, const _Float16* __restrict__ WT,
                const float* __restrict__ bias, const float* __restrict__ residual,
                _Float16* __restrict__ dst16, float* __restrict__ dst32,
                int M, int N, int K) {
  __shared__ _Float16 sA[128][40];   // [m][k]  rows 80B apart (16B aligned)
  __shared__ _Float16 sBt[128][40];  // [n][k]
  const int tid  = threadIdx.x;
  const int lane = tid & 31;
  const int wv   = tid >> 5;
  const int m0   = blockIdx.x * 128;
  const int n0   = blockIdx.y * 128;
  const int wrow = (wv >> 1) * 32;
  const int wcol = (wv & 1) * 64;

  const int sr = tid >> 2;            // 0..63
  const int sk = (tid & 3) << 3;      // 0,8,16,24

  v8f zero = {};
  v8f acc[2][4];
#pragma unroll
  for (int i = 0; i < 2; ++i)
#pragma unroll
    for (int j = 0; j < 4; ++j) acc[i][j] = zero;

  for (int k0 = 0; k0 < K; k0 += 32) {
    __syncthreads();
    const unsigned aoff = (unsigned)((m0 + sr) * K + k0 + sk);
    const unsigned boff = (unsigned)((n0 + sr) * K + k0 + sk);
    const unsigned step = (unsigned)(64 * K);
    v8h a0 = *(const v8h*)(A + aoff);
    v8h a1 = *(const v8h*)(A + aoff + step);
    v8h b0 = *(const v8h*)(WT + boff);
    v8h b1 = *(const v8h*)(WT + boff + step);
    if (k0 + 32 < K) {  // hint next tiles into L2 (global_prefetch_b8)
      __builtin_prefetch(A + aoff + 32, 0, 1);
      __builtin_prefetch(WT + boff + 32, 0, 1);
    }
    *(v8h*)&sA[sr][sk]       = a0;
    *(v8h*)&sA[sr + 64][sk]  = a1;
    *(v8h*)&sBt[sr][sk]      = b0;
    *(v8h*)&sBt[sr + 64][sk] = b1;
    __syncthreads();

    v16h af[2], bf[4];
#pragma unroll
    for (int i = 0; i < 2; ++i) af[i] = frag_a(&sA[wrow + i * 16][0], 40, lane);
#pragma unroll
    for (int j = 0; j < 4; ++j) bf[j] = frag_bt(&sBt[wcol + j * 16][0], 40, lane);
#pragma unroll
    for (int i = 0; i < 2; ++i)
#pragma unroll
      for (int j = 0; j < 4; ++j)
        acc[i][j] = wmma_16x16x32(af[i], bf[j], acc[i][j]);
  }

#pragma unroll
  for (int i = 0; i < 2; ++i)
#pragma unroll
    for (int j = 0; j < 4; ++j) {
      const int colg = n0 + wcol + j * 16 + (lane & 15);
      const float bv = bias[colg];
      const int rb = m0 + wrow + i * 16 + ((lane >> 4) << 3);
#pragma unroll
      for (int r = 0; r < 8; ++r) {
        const int rowg = rb + r;
        const float val = acc[i][j][r] + bv;
        if (OMODE == 0) {
          dst16[(size_t)rowg * N + colg] = (_Float16)val;
        } else if (OMODE == 1) {
          const int bh = rowg / 48, w = rowg - bh * 48;
          dst16[((size_t)bh * N + colg) * 48 + w] = (_Float16)val;
        } else {
          const size_t idx = (size_t)rowg * N + colg;
          dst32[idx] = val + residual[idx];
        }
      }
    }
}

// ---------------------------------------------------------------------------
// attn[bh, c, k] = scale * sum_w q[bh, w, c] * k[bh, w, k]
// qT/kT are [bh][512][48] (w contiguous): both tiles stage as b128 copies.
// w=48 zero-padded to 64 (two K=32 steps).  f32 out (pre-softmax).
// ---------------------------------------------------------------------------
__global__ void __launch_bounds__(256)
attn_qk_kernel(const _Float16* __restrict__ qT, const _Float16* __restrict__ kT,
               float* __restrict__ attn, float scale) {
  __shared__ _Float16 sA[128][40];   // [c_local][w_local]
  __shared__ _Float16 sBt[128][40];  // [k_local][w_local]
  const int tid  = threadIdx.x;
  const int lane = tid & 31;
  const int wv   = tid >> 5;
  const int bh   = blockIdx.y;
  const int c0   = (blockIdx.x >> 2) * 128;
  const int n0   = (blockIdx.x & 3) * 128;
  const _Float16* qb = qT + (size_t)bh * 512 * 48;
  const _Float16* kb = kT + (size_t)bh * 512 * 48;
  float* ab = attn + (size_t)bh * 512 * 512;
  const int wrow = (wv >> 1) * 32;
  const int wcol = (wv & 1) * 64;

  const int sr = tid >> 2;            // 0..63
  const int sw = (tid & 3) << 3;      // 0,8,16,24

  v8f zero = {};
  v8f acc[2][4];
#pragma unroll
  for (int i = 0; i < 2; ++i)
#pragma unroll
    for (int j = 0; j < 4; ++j) acc[i][j] = zero;

#pragma unroll
  for (int w0 = 0; w0 < 64; w0 += 32) {
    __syncthreads();
    const int w = w0 + sw;
    v8h a0 = {}, a1 = {}, b0 = {}, b1 = {};
    if (w < 48) {
      const unsigned qoff = (unsigned)((c0 + sr) * 48 + w);
      const unsigned koff = (unsigned)((n0 + sr) * 48 + w);
      a0 = *(const v8h*)(qb + qoff);
      a1 = *(const v8h*)(qb + qoff + 64 * 48);
      b0 = *(const v8h*)(kb + koff);
      b1 = *(const v8h*)(kb + koff + 64 * 48);
    }
    *(v8h*)&sA[sr][sw]       = a0;
    *(v8h*)&sA[sr + 64][sw]  = a1;
    *(v8h*)&sBt[sr][sw]      = b0;
    *(v8h*)&sBt[sr + 64][sw] = b1;
    __syncthreads();

    v16h af[2], bf[4];
#pragma unroll
    for (int i = 0; i < 2; ++i) af[i] = frag_a(&sA[wrow + i * 16][0], 40, lane);
#pragma unroll
    for (int j = 0; j < 4; ++j) bf[j] = frag_bt(&sBt[wcol + j * 16][0], 40, lane);
#pragma unroll
    for (int i = 0; i < 2; ++i)
#pragma unroll
      for (int j = 0; j < 4; ++j)
        acc[i][j] = wmma_16x16x32(af[i], bf[j], acc[i][j]);
  }

#pragma unroll
  for (int i = 0; i < 2; ++i)
#pragma unroll
    for (int j = 0; j < 4; ++j) {
      const int colg = n0 + wcol + j * 16 + (lane & 15);
      const int rb = c0 + wrow + i * 16 + ((lane >> 4) << 3);
#pragma unroll
      for (int r = 0; r < 8; ++r)
        ab[(size_t)(rb + r) * 512 + colg] = acc[i][j][r] * scale;
    }
}

// ---------------------------------------------------------------------------
// In-place softmax over axis c of attn[bh, c(512), k(512)] (columns).
// ---------------------------------------------------------------------------
__global__ void __launch_bounds__(256)
softmax_c_kernel(float* __restrict__ attn) {
  const int bh = blockIdx.x;
  const int k0 = blockIdx.y * 64;
  float* base = attn + (size_t)bh * 512 * 512 + k0;
  const int kx = threadIdx.x & 63;
  const int cy = threadIdx.x >> 6;   // 0..3
  __shared__ float red[4][64];

  float m = -3.402823466e38f;
#pragma unroll 8
  for (int c = cy; c < 512; c += 4) m = fmaxf(m, base[(size_t)c * 512 + kx]);
  red[cy][kx] = m;
  __syncthreads();
  m = fmaxf(fmaxf(red[0][kx], red[1][kx]), fmaxf(red[2][kx], red[3][kx]));
  __syncthreads();

  float s = 0.f;
#pragma unroll 8
  for (int c = cy; c < 512; c += 4) s += expf(base[(size_t)c * 512 + kx] - m);
  red[cy][kx] = s;
  __syncthreads();
  s = red[0][kx] + red[1][kx] + red[2][kx] + red[3][kx];
  const float invs = 1.0f / s;

#pragma unroll 8
  for (int c = cy; c < 512; c += 4) {
    const size_t idx = (size_t)c * 512 + kx;
    base[idx] = expf(base[idx] - m) * invs;
  }
}

// ---------------------------------------------------------------------------
// out1[bh, w, k] = sum_d v[bh, w, d] * attn[bh, d, k]
// M=w(48 -> 3 tiles), N=k(512), K=d(512).  attn (f32) transposed+converted
// to f16 while staging so B fragments are contiguous.
// ---------------------------------------------------------------------------
__global__ void __launch_bounds__(256)
attn_v_kernel(const _Float16* __restrict__ v, const float* __restrict__ attn,
              _Float16* __restrict__ out1) {
  __shared__ _Float16 sA[48][40];    // [w][d_local]
  __shared__ _Float16 sBt[128][40];  // [k_local][d_local]
  const int tid  = threadIdx.x;
  const int lane = tid & 31;
  const int wv   = tid >> 5;
  const int bh   = blockIdx.y;
  const int n0   = blockIdx.x * 128;
  const _Float16* vb = v + (size_t)bh * 48 * 512;
  const float* ab = attn + (size_t)bh * 512 * 512;

  const int srA = tid >> 2, skA = (tid & 3) << 3;   // v tile (tid < 192)
  const int srB = tid >> 5, skB = (tid & 31) << 2;  // attn tile

  v8f zero = {};
  v8f acc[3] = {zero, zero, zero};

  for (int d0 = 0; d0 < 512; d0 += 32) {
    __syncthreads();
    // issue all global loads first
    v8h va = {};
    if (tid < 192) va = *(const v8h*)(vb + (unsigned)(srA * 512 + d0 + skA));
    const unsigned aoff = (unsigned)((d0 + srB) * 512 + n0 + skB);
    v4f f0 = *(const v4f*)(ab + aoff);
    v4f f1 = *(const v4f*)(ab + aoff + 8 * 512);
    v4f f2 = *(const v4f*)(ab + aoff + 16 * 512);
    v4f f3 = *(const v4f*)(ab + aoff + 24 * 512);
    if (tid < 192) *(v8h*)&sA[srA][skA] = va;
#pragma unroll
    for (int q = 0; q < 4; ++q) {
      v4f f = (q == 0) ? f0 : (q == 1) ? f1 : (q == 2) ? f2 : f3;
      const int d = srB + q * 8;
      sBt[skB + 0][d] = (_Float16)f[0];
      sBt[skB + 1][d] = (_Float16)f[1];
      sBt[skB + 2][d] = (_Float16)f[2];
      sBt[skB + 3][d] = (_Float16)f[3];
    }
    __syncthreads();

    v16h bf = frag_bt(&sBt[wv * 16][0], 40, lane);
#pragma unroll
    for (int i = 0; i < 3; ++i) {
      v16h af = frag_a(&sA[i * 16][0], 40, lane);
      acc[i] = wmma_16x16x32(af, bf, acc[i]);
    }
  }

#pragma unroll
  for (int i = 0; i < 3; ++i)
#pragma unroll
    for (int r = 0; r < 8; ++r) {
      const int row = i * 16 + ((lane >> 4) << 3) + r;       // 0..47
      const int col = n0 + wv * 16 + (lane & 15);
      out1[((size_t)bh * 48 + row) * 512 + col] = (_Float16)acc[i][r];
    }
}

// ---------------------------------------------------------------------------
// Host-side orchestration.
// ---------------------------------------------------------------------------
extern "C" void kernel_launch(void* const* d_in, const int* in_sizes, int n_in,
                              void* d_out, int out_size, void* d_ws, size_t ws_size,
                              hipStream_t stream) {
  (void)in_sizes; (void)n_in; (void)out_size;

  const float* hidden = (const float*)d_in[0];
  const float* enc    = (const float*)d_in[1];
  const float* ln_g   = (const float*)d_in[2];
  const float* ln_b   = (const float*)d_in[3];
  const float* eln_g  = (const float*)d_in[4];
  const float* eln_b  = (const float*)d_in[5];
  const float* Wq     = (const float*)d_in[6];
  const float* bq     = (const float*)d_in[7];
  const float* Wk     = (const float*)d_in[8];
  const float* bk     = (const float*)d_in[9];
  const float* Wv     = (const float*)d_in[10];
  const float* bv     = (const float*)d_in[11];
  const float* Wo     = (const float*)d_in[12];
  const float* bo     = (const float*)d_in[13];

  constexpr int HID = 512, ENC = 768;
  constexpr int NBH  = 8 * 48;        // 384 (b,h) batches
  constexpr int NTOK = NBH * 48;      // 18432 tokens

  _Float16* ws = (_Float16*)d_ws;
  size_t off = 0;
  _Float16* x16   = ws + off; off += (size_t)NTOK * HID;
  _Float16* e16   = ws + off; off += (size_t)NTOK * ENC;
  _Float16* WqT16 = ws + off; off += (size_t)HID * HID;
  _Float16* WkT16 = ws + off; off += (size_t)ENC * HID;
  _Float16* WvT16 = ws + off; off += (size_t)ENC * HID;
  _Float16* WoT16 = ws + off; off += (size_t)HID * HID;
  _Float16* qT16  = ws + off; off += (size_t)NTOK * HID;   // [bh][c][w]
  _Float16* kT16  = ws + off; off += (size_t)NTOK * HID;   // [bh][k][w]
  _Float16* v16   = ws + off; off += (size_t)NTOK * HID;   // [bh][w][d]
  _Float16* o16   = ws + off; off += (size_t)NTOK * HID;
  if (ws_size < off * sizeof(_Float16)) return;  // not enough scratch

  float* outp  = (float*)d_out;                      // [8,48,48,512]
  float* attnp = outp + (size_t)NTOK * HID;          // [8,48,512,512]

  const float scale = sqrtf(1.0f / 48.0f);

  // 1) layernorms -> f16 activations
  ln_cast_kernel<HID><<<NTOK, 256, 0, stream>>>(hidden, ln_g, ln_b, x16);
  ln_cast_kernel<ENC><<<NTOK, 256, 0, stream>>>(enc, eln_g, eln_b, e16);

  // 2) weights -> transposed f16
  cvt_transpose_kernel<<<dim3(HID / 256, HID), 256, 0, stream>>>(Wq, WqT16, HID, HID);
  cvt_transpose_kernel<<<dim3(HID / 256, ENC), 256, 0, stream>>>(Wk, WkT16, ENC, HID);
  cvt_transpose_kernel<<<dim3(HID / 256, ENC), 256, 0, stream>>>(Wv, WvT16, ENC, HID);
  cvt_transpose_kernel<<<dim3(HID / 256, HID), 256, 0, stream>>>(Wo, WoT16, HID, HID);

  // 3) q / k / v projections (WMMA).  q,k stored batch-transposed (OMODE 1).
  dim3 gproj(NTOK / 128, HID / 128);
  gemm_f16_kernel<1><<<gproj, 256, 0, stream>>>(x16, WqT16, bq, nullptr, qT16,
                                                nullptr, NTOK, HID, HID);
  gemm_f16_kernel<1><<<gproj, 256, 0, stream>>>(e16, WkT16, bk, nullptr, kT16,
                                                nullptr, NTOK, HID, ENC);
  gemm_f16_kernel<0><<<gproj, 256, 0, stream>>>(e16, WvT16, bv, nullptr, v16,
                                                nullptr, NTOK, HID, ENC);

  // 4) attn = scale * q^T k   (written straight into d_out's attn segment)
  attn_qk_kernel<<<dim3(16, NBH), 256, 0, stream>>>(qT16, kT16, attnp, scale);

  // 5) softmax over channel axis, in place
  softmax_c_kernel<<<dim3(NBH, 8), 256, 0, stream>>>(attnp);

  // 6) out1 = v @ attn (attn f32 -> f16 + transpose on the fly)
  attn_v_kernel<<<dim3(4, NBH), 256, 0, stream>>>(v16, attnp, o16);

  // 7) out = out1 @ Wo + bo + hidden  (f32 to d_out)
  gemm_f16_kernel<2><<<gproj, 256, 0, stream>>>(o16, WoT16, bo, hidden, nullptr,
                                                outp, NTOK, HID, HID);
}
